// mambaEncoder_87136296501516
// MI455X (gfx1250) — matrix operand order repaired and checked by reading, not compiled, not run on hardware
//
#include <hip/hip_runtime.h>

typedef unsigned short u16;
typedef __attribute__((ext_vector_type(16))) __bf16 v16bf;
typedef __attribute__((ext_vector_type(8)))  float  v8f;

#define B_SZ    16
#define L_SEQ   2048
#define D_MODEL 256
#define D_INNER 512
#define D_STATE 16
#define DT_RANK 16
#define M_TOT   (B_SZ * L_SEQ)   // 32768
#define K_MAX   512              // largest GEMM K (fits 64xK bf16 slab in LDS)

__device__ __forceinline__ u16 f2bf(float f) {
    unsigned int u = __float_as_uint(f);
    unsigned int r = u + 0x7FFFu + ((u >> 16) & 1u);  // round-to-nearest-even
    return (u16)(r >> 16);
}

// ---------------------------------------------------------------------------
// Weight fp32 -> bf16 with N padded to Npad (zero rows beyond N)
// ---------------------------------------------------------------------------
__global__ __launch_bounds__(256)
void convert_w_kernel(const float* __restrict__ src, u16* __restrict__ dst,
                      int N, int K, int Npad) {
    size_t i = (size_t)blockIdx.x * 256 + threadIdx.x;
    if (i >= (size_t)Npad * K) return;
    int n = (int)(i / K), k = (int)(i % K);
    float v = (n < N) ? src[(size_t)n * K + k] : 0.0f;
    dst[i] = f2bf(v);
}

// ---------------------------------------------------------------------------
// LayerNorm over D=256 (one block per token), optional residual add.
// ---------------------------------------------------------------------------
__global__ __launch_bounds__(256)
void ln_kernel(const float* __restrict__ in, const float* __restrict__ resid,
               const float* __restrict__ g, const float* __restrict__ bt,
               float* __restrict__ z_out, u16* __restrict__ out_bf) {
    __shared__ float red[256];
    int tid = threadIdx.x;
    size_t i = (size_t)blockIdx.x * 256 + tid;
    float v = in[i];
    if (resid) v += resid[i];
    if (z_out) z_out[i] = v;
    red[tid] = v; __syncthreads();
    for (int s = 128; s > 0; s >>= 1) { if (tid < s) red[tid] += red[tid + s]; __syncthreads(); }
    float mu = red[0] * (1.0f / 256.0f);
    __syncthreads();
    float dv = v - mu;
    red[tid] = dv * dv; __syncthreads();
    for (int s = 128; s > 0; s >>= 1) { if (tid < s) red[tid] += red[tid + s]; __syncthreads(); }
    float rstd = rsqrtf(red[0] * (1.0f / 256.0f) + 1e-5f);
    out_bf[i] = f2bf(dv * rstd * g[tid] + bt[tid]);
}

// ---------------------------------------------------------------------------
// bf16 WMMA GEMM:  C[M,Npad] = A[M,K] * W[Npad,K]^T (+bias +resid)
//   Block = 256 threads = 8 waves; block tile = 256(M) x 64(N).
//   Weight slab W[n0:n0+64, 0:K] staged in LDS once (<=64KB).
//   Wave tile = 32(M) x 64(N) -> 8 f32 accumulators.
//   k-loop unrolled by 64 with ping-pong A register buffers (no copies);
//   all 4 B fragments of a k-step loaded from LDS before the 8 WMMAs so
//   there is a single dscnt wait per 8 matrix ops.
// A-frag (16-bit A 16x32): lane m%16, K chunks {koff, koff+16}, koff=(lane>=16)*8
// B-frag (16-bit B 32x16): lane n%16, K chunks {koff, koff+16}, koff=(lane>=16)*8
// ---------------------------------------------------------------------------
__global__ __launch_bounds__(256)
void gemm_bf16_wmma_kernel(const u16* __restrict__ A, const u16* __restrict__ W,
                           float* __restrict__ C, const float* __restrict__ bias,
                           const float* __restrict__ resid, int K, int Npad) {
    __shared__ u16 wlds[64 * K_MAX];   // 64 KB @ K=512

    const int wave = threadIdx.x >> 5;
    const int lane = threadIdx.x & 31;
    const int m0 = (blockIdx.x * 8 + wave) * 32;
    const int n0 = blockIdx.y * 64;

    // ---- stage 64 x K weight slab into LDS (rows n0..n0+63 are contiguous)
    {
        const u16* slab = W + (size_t)n0 * K;
        const int total = 64 * K;
        for (int idx = threadIdx.x * 8; idx < total; idx += 256 * 8)
            *(uint4*)&wlds[idx] = *(const uint4*)(slab + idx);
    }
    __syncthreads();

    const int row   = lane & 15;
    const int koffA = (lane >> 4) << 3;   // 0 or 8
    const int koffB = (lane >> 4) << 3;   // 0 or 8

    const u16* apLo = A + (size_t)(m0 + row) * K + koffA;
    const u16* apHi = apLo + (size_t)16 * K;
    const u16* ldsBase = &wlds[(size_t)row * K + koffB];

    v8f acc[8] = {};
    v16bf a0L, a0H, a1L, a1H;

    // preload k = 0 into buffer 0
    ((uint4*)&a0L)[0] = *(const uint4*)(apLo);
    ((uint4*)&a0L)[1] = *(const uint4*)(apLo + 16);
    ((uint4*)&a0H)[0] = *(const uint4*)(apHi);
    ((uint4*)&a0H)[1] = *(const uint4*)(apHi + 16);

    // one 32-wide k-step: consume (aL,aH) at k, load (aLn,aHn) at k+32
    auto kstep = [&](int k, const v16bf& aL, const v16bf& aH,
                     v16bf& aLn, v16bf& aHn, bool loadNext) {
        if (loadNext) {
            ((uint4*)&aLn)[0] = *(const uint4*)(apLo + k + 32);
            ((uint4*)&aLn)[1] = *(const uint4*)(apLo + k + 48);
            ((uint4*)&aHn)[0] = *(const uint4*)(apHi + k + 32);
            ((uint4*)&aHn)[1] = *(const uint4*)(apHi + k + 48);
            __builtin_prefetch(apLo + k + 64, 0, 1);   // global_prefetch_b8
        }
        v16bf b[4];
#pragma unroll
        for (int t = 0; t < 4; ++t) {
            const u16* lp = ldsBase + (size_t)t * 16 * K + k;
            ((uint4*)&b[t])[0] = *(const uint4*)(lp);       // ds_load_b128
            ((uint4*)&b[t])[1] = *(const uint4*)(lp + 16);  // ds_load_b128
        }
#pragma unroll
        for (int t = 0; t < 4; ++t) {
            acc[t]     = __builtin_amdgcn_wmma_f32_16x16x32_bf16(
                false, aL, false, b[t], (short)0, acc[t],     false, false);
            acc[4 + t] = __builtin_amdgcn_wmma_f32_16x16x32_bf16(
                false, aH, false, b[t], (short)0, acc[4 + t], false, false);
        }
    };

    // K is a multiple of 64: ping-pong without register copies
    for (int k = 0; k < K; k += 64) {
        kstep(k,      a0L, a0H, a1L, a1H, true);
        kstep(k + 32, a1L, a1H, a0L, a0H, (k + 64) < K);
    }

    // ---- epilogue: C/D layout -> lane%16 = col, VGPR r = row (+8 for lane>=16)
    const int ncol = n0 + (lane & 15);
    const int rofs = (lane >> 4) << 3;
#pragma unroll
    for (int rg = 0; rg < 2; ++rg) {
#pragma unroll
        for (int t = 0; t < 4; ++t) {
            int nc = ncol + t * 16;
#pragma unroll
            for (int r = 0; r < 8; ++r) {
                float v = acc[rg * 4 + t][r];
                size_t idx = (size_t)(m0 + rg * 16 + rofs + r) * Npad + nc;
                if (bias)  v += bias[nc];
                if (resid) v += resid[idx];
                C[idx] = v;
            }
        }
    }
}

// ---------------------------------------------------------------------------
// Causal depthwise conv (D_CONV=4) + SiLU.  xz row = [xi(512) | z(512)].
// ---------------------------------------------------------------------------
__global__ __launch_bounds__(256)
void conv_silu_kernel(const float* __restrict__ xz, const float* __restrict__ cw,
                      const float* __restrict__ cb, float* __restrict__ u,
                      u16* __restrict__ ubf) {
    size_t gid = (size_t)blockIdx.x * 256 + threadIdx.x;   // token*512 + d
    int d = (int)(gid & (D_INNER - 1));
    size_t tok = gid >> 9;
    int l = (int)(tok & (L_SEQ - 1));
    float acc = cb[d];
#pragma unroll
    for (int j = 0; j < 4; ++j) {
        int lj = l - 3 + j;
        if (lj >= 0) acc += xz[(tok - 3 + j) * 1024 + d] * cw[d * 4 + j];
    }
    float s = acc / (1.0f + __expf(-acc));   // silu
    u[gid] = s;
    ubf[gid] = f2bf(s);
}

// ---------------------------------------------------------------------------
// delta = softplus(dt @ dt_proj_w^T + dt_proj_b); dt = x_dbl[:, :16]
// ---------------------------------------------------------------------------
__global__ __launch_bounds__(256)
void delta_kernel(const float* __restrict__ xdbl, const float* __restrict__ dtw,
                  const float* __restrict__ dtb, float* __restrict__ delta) {
    size_t gid = (size_t)blockIdx.x * 256 + threadIdx.x;
    int d = (int)(gid & (D_INNER - 1));
    size_t tok = gid >> 9;
    const float* xr = xdbl + tok * 64;
    float acc = dtb[d];
#pragma unroll
    for (int r = 0; r < DT_RANK; ++r) acc += xr[r] * dtw[d * DT_RANK + r];
    delta[gid] = (acc > 20.0f) ? acc : log1pf(__expf(acc));
}

// ---------------------------------------------------------------------------
// Selective scan: one block per batch b (512 threads = one channel each,
// 16 states in registers).  B/C staged into LDS in 64-step chunks.
// ---------------------------------------------------------------------------
__global__ __launch_bounds__(512)
void scan_kernel(const float* __restrict__ xdbl, const float* __restrict__ delta,
                 const float* __restrict__ u, const float* __restrict__ A_log,
                 float* __restrict__ ys) {
    __shared__ float bc[64 * 32];          // [t][0:16)=B, [t][16:32)=C
    const int b = blockIdx.x;
    const int d = threadIdx.x;
    float Ar[D_STATE], h[D_STATE];
#pragma unroll
    for (int n = 0; n < D_STATE; ++n) {
        Ar[n] = -__expf(A_log[d * D_STATE + n]);
        h[n] = 0.0f;
    }
    for (int t0 = 0; t0 < L_SEQ; t0 += 64) {
        for (int i = threadIdx.x; i < 64 * 32; i += 512) {
            int t = i >> 5, j = i & 31;
            bc[i] = xdbl[((size_t)b * L_SEQ + t0 + t) * 64 + 16 + j]; // cols 16..47 = B|C
        }
        __syncthreads();
        for (int t = 0; t < 64; ++t) {
            size_t tok = (size_t)b * L_SEQ + t0 + t;
            float dlt = delta[tok * D_INNER + d];
            float du  = dlt * u[tok * D_INNER + d];
            const float* Bt = &bc[t * 32];
            const float* Ct = Bt + 16;
            float y = 0.0f;
#pragma unroll
            for (int n = 0; n < D_STATE; ++n) {
                h[n] = __expf(dlt * Ar[n]) * h[n] + du * Bt[n];
                y += h[n] * Ct[n];
            }
            ys[tok * D_INNER + d] = y;
        }
        __syncthreads();
    }
}

// ---------------------------------------------------------------------------
// y2 = (ys + u*Dp) * silu(z)  -> bf16 for out_proj GEMM
// ---------------------------------------------------------------------------
__global__ __launch_bounds__(256)
void gate_kernel(const float* __restrict__ ys, const float* __restrict__ u,
                 const float* __restrict__ Dp, const float* __restrict__ xz,
                 u16* __restrict__ y2bf) {
    size_t gid = (size_t)blockIdx.x * 256 + threadIdx.x;
    int d = (int)(gid & (D_INNER - 1));
    size_t tok = gid >> 9;
    float z = xz[tok * 1024 + 512 + d];
    float y = (ys[gid] + u[gid] * Dp[d]) * (z / (1.0f + __expf(-z)));
    y2bf[gid] = f2bf(y);
}

// ---------------------------------------------------------------------------
extern "C" void kernel_launch(void* const* d_in, const int* in_sizes, int n_in,
                              void* d_out, int out_size, void* d_ws, size_t ws_size,
                              hipStream_t stream) {
    const float* x        = (const float*)d_in[0];
    const float* ln_g     = (const float*)d_in[1];
    const float* ln_b     = (const float*)d_in[2];
    const float* in_proj  = (const float*)d_in[3];   // [1024,256]
    const float* conv_w   = (const float*)d_in[4];   // [512,1,4]
    const float* conv_b   = (const float*)d_in[5];
    const float* x_proj   = (const float*)d_in[6];   // [48,512]
    const float* dt_w     = (const float*)d_in[7];   // [512,16]
    const float* dt_b     = (const float*)d_in[8];
    const float* A_log    = (const float*)d_in[9];   // [512,16]
    const float* Dp       = (const float*)d_in[10];
    const float* out_proj = (const float*)d_in[11];  // [256,512]
    const float* ln1_g    = (const float*)d_in[12];
    const float* ln1_b    = (const float*)d_in[13];
    const float* mlp_w    = (const float*)d_in[14];  // [256,256]
    const float* mlp_b    = (const float*)d_in[15];

    char* wsp = (char*)d_ws;
    size_t off = 0;
    auto alloc = [&](size_t bytes) -> void* {
        void* p = wsp + off;
        off = (off + bytes + 255) & ~(size_t)255;
        return p;
    };
    const size_t M = M_TOT;
    u16*   wpA   = (u16*)  alloc((size_t)1024 * 256 * 2);
    u16*   wpX   = (u16*)  alloc((size_t)64   * 512 * 2);
    u16*   wpO   = (u16*)  alloc((size_t)256  * 512 * 2);
    u16*   wpM   = (u16*)  alloc((size_t)256  * 256 * 2);
    u16*   xnbf  = (u16*)  alloc(M * 256 * 2);     // reused for LN2 output
    float* xz    = (float*)alloc(M * 1024 * 4);    // prefix reused as z2 later
    float* u     = (float*)alloc(M * 512 * 4);
    u16*   ubf   = (u16*)  alloc(M * 512 * 2);     // reused for gated y bf16
    float* xdbl  = (float*)alloc(M * 64 * 4);      // padded x_dbl (48 -> 64)
    float* delta = (float*)alloc(M * 512 * 4);
    float* ys    = (float*)alloc(M * 512 * 4);
    float* mout  = (float*)alloc(M * 256 * 4);
    float* z2    = xz;                             // xz dead after gate_kernel

    // 1) weights -> bf16 (x_proj padded 48->64 with zero rows)
    convert_w_kernel<<<(1024 * 256 + 255) / 256, 256, 0, stream>>>(in_proj,  wpA, 1024, 256, 1024);
    convert_w_kernel<<<(64 * 512 + 255) / 256,   256, 0, stream>>>(x_proj,   wpX, 48,   512, 64);
    convert_w_kernel<<<(256 * 512 + 255) / 256,  256, 0, stream>>>(out_proj, wpO, 256,  512, 256);
    convert_w_kernel<<<(256 * 256 + 255) / 256,  256, 0, stream>>>(mlp_w,    wpM, 256,  256, 256);

    // 2) pre-norm LN -> bf16
    ln_kernel<<<M, 256, 0, stream>>>(x, nullptr, ln_g, ln_b, nullptr, xnbf);

    // 3) in_proj GEMM: [M,256] x [1024,256]^T -> xz [M,1024]
    gemm_bf16_wmma_kernel<<<dim3(M / 256, 1024 / 64), 256, 0, stream>>>(
        xnbf, wpA, xz, nullptr, nullptr, 256, 1024);

    // 4) causal depthwise conv + silu -> u (fp32 + bf16)
    conv_silu_kernel<<<(M * 512) / 256, 256, 0, stream>>>(xz, conv_w, conv_b, u, ubf);

    // 5) x_proj GEMM: [M,512] x [64,512]^T -> x_dbl [M,64] (cols 48..63 zero)
    gemm_bf16_wmma_kernel<<<dim3(M / 256, 1), 256, 0, stream>>>(
        ubf, wpX, xdbl, nullptr, nullptr, 512, 64);

    // 6) delta = softplus(dt @ dt_proj^T + b)
    delta_kernel<<<(M * 512) / 256, 256, 0, stream>>>(xdbl, dt_w, dt_b, delta);

    // 7) selective scan (sequential in L, parallel over B x D_INNER)
    scan_kernel<<<B_SZ, 512, 0, stream>>>(xdbl, delta, u, A_log, ys);

    // 8) gate: y = (ys + u*D) * silu(z) -> bf16
    gate_kernel<<<(M * 512) / 256, 256, 0, stream>>>(ys, u, Dp, xz, ubf);

    // 9) out_proj GEMM: [M,512] x [256,512]^T -> mout [M,256]
    gemm_bf16_wmma_kernel<<<dim3(M / 256, 256 / 64), 256, 0, stream>>>(
        ubf, wpO, mout, nullptr, nullptr, 512, 256);

    // 10) z2 = mout + x (residual); LN1(z2) -> bf16
    ln_kernel<<<M, 256, 0, stream>>>(mout, x, ln1_g, ln1_b, z2, xnbf);

    // 11) MLP GEMM with fused bias + residual(z2) -> d_out
    gemm_bf16_wmma_kernel<<<dim3(M / 256, 256 / 64), 256, 0, stream>>>(
        xnbf, wpM, (float*)d_out, mlp_b, z2, 256, 256);
}
